// Seq2SeqModel_52149492908602
// MI455X (gfx1250) — compile-verified, matrix operand order checked by reading, not run on hardware
//
#include <hip/hip_runtime.h>

// ---------------------------------------------------------------------------
// GRU seq2seq on MI455X (gfx1250): bf16 WMMA GEMMs, f32 accumulate/state.
// Wave tile = 32Mx64N (2 A-frags x 4 B-frags -> 8 v_wmma per k-step of 32),
// ~21 flops/byte from L2 vs 8 for a single-tile wave.
// ---------------------------------------------------------------------------

typedef __attribute__((ext_vector_type(16))) __bf16 v16bf;
typedef __attribute__((ext_vector_type(8)))  __bf16 v8bf;
typedef __attribute__((ext_vector_type(8)))  float  v8f;

constexpr int BATCH  = 1024;
constexpr int H      = 1024;
constexpr int G3     = 3 * H;      // 3072
constexpr int IN     = 55;
constexpr int KP     = 64;         // input size padded to WMMA K multiple
constexpr int SRC_T  = 50;
constexpr int TGT_T  = 25;
constexpr int WPB    = 4;          // waves per block (wave32 -> 128 threads)

__device__ __forceinline__ float sigmoidf_(float x) {
    return 1.0f / (1.0f + __expf(-x));
}

// ---------------------------------------------------------------------------
// One-time weight conversion: f32 -> bf16, zero padding K=55->64 (W_ih) and
// rows 55->64 (fc_w).
// ---------------------------------------------------------------------------
__global__ void prep_weights(const float* __restrict__ Wih,   // [3072][55]
                             const float* __restrict__ Whh,   // [3072][1024]
                             const float* __restrict__ fcw,   // [55][1024]
                             __bf16* __restrict__ Wih_b,      // [3072][64]
                             __bf16* __restrict__ Whh_b,      // [3072][1024]
                             __bf16* __restrict__ fcw_b) {    // [64][1024]
    const int total = G3 * KP + G3 * H + KP * H;
    for (int i = blockIdx.x * blockDim.x + threadIdx.x; i < total;
         i += gridDim.x * blockDim.x) {
        if (i < G3 * KP) {
            int r = i / KP, c = i % KP;
            Wih_b[i] = (c < IN) ? (__bf16)Wih[r * IN + c] : (__bf16)0.0f;
        } else if (i < G3 * KP + G3 * H) {
            int j = i - G3 * KP;
            Whh_b[j] = (__bf16)Whh[j];
        } else {
            int j = i - G3 * KP - G3 * H;
            int r = j / H;
            fcw_b[j] = (r < IN) ? (__bf16)fcw[j] : (__bf16)0.0f;
        }
    }
}

__global__ void init_h(float* __restrict__ h, __bf16* __restrict__ hb) {
    int i = blockIdx.x * blockDim.x + threadIdx.x;
    if (i < BATCH * H) { h[i] = 0.0f; hb[i] = (__bf16)0.0f; }
}

// Pad/convert one [B,55] f32 time-slice into [B,64] bf16 (zeros in pad lanes).
__global__ void conv_x(const float* __restrict__ src, int row_stride,
                       __bf16* __restrict__ xb) {
    int i = blockIdx.x * blockDim.x + threadIdx.x;
    if (i >= BATCH * KP) return;
    int b = i / KP, c = i % KP;
    xb[i] = (c < IN) ? (__bf16)src[b * row_stride + c] : (__bf16)0.0f;
}

// ---------------------------------------------------------------------------
// Fragment packing per ISA 16-bit operand layout:
//   lanes 0-15 : row = lane,    K halves {0..7, 16..23} (+k0)
//   lanes 16-31: row = lane-16, K halves {8..15, 24..31} (+k0)
// -> two contiguous 16B chunks per fragment (global_load_b128 pairs).
// ---------------------------------------------------------------------------
__device__ __forceinline__ v16bf pack_frag(const __bf16* p, int k0) {
    v8bf lo = *(const v8bf*)(p + k0);
    v8bf hi = *(const v8bf*)(p + k0 + 16);
    v16bf f;
#pragma unroll
    for (int i = 0; i < 8; ++i) { f[i] = lo[i]; f[i + 8] = hi[i]; }
    return f;
}

// ---------------------------------------------------------------------------
// out[M,N] = A[M,K](bf16) * W[N,K]^T(bf16) + bias[N], f32 accumulate.
// Wave tile: 32Mx64N. tilesN4 = N/64.
// ---------------------------------------------------------------------------
__global__ void gemm_bf16_wmma(const __bf16* __restrict__ A, int lda,
                               const __bf16* __restrict__ W, int ldw,
                               const float* __restrict__ bias,
                               float* __restrict__ out, int ldo,
                               int tilesN4, int K) {
    const int wave = threadIdx.x >> 5;
    const int lane = threadIdx.x & 31;
    const int tile = blockIdx.x * (blockDim.x >> 5) + wave;
    const int m0 = (tile / tilesN4) * 32;
    const int n0 = (tile % tilesN4) * 64;
    const int r  = lane & 15;
    const int kb = (lane < 16) ? 0 : 8;

    const __bf16* ap0 = A + (size_t)(m0 + r) * lda + kb;
    const __bf16* ap1 = ap0 + (size_t)16 * lda;
    const __bf16* bp0 = W + (size_t)(n0 + r) * ldw + kb;
    const __bf16* bp1 = bp0 + (size_t)16 * ldw;
    const __bf16* bp2 = bp0 + (size_t)32 * ldw;
    const __bf16* bp3 = bp0 + (size_t)48 * ldw;

    v8f acc[2][4] = {};
#pragma unroll 1
    for (int k0 = 0; k0 < K; k0 += 32) {
        v16bf a0 = pack_frag(ap0, k0);
        v16bf a1 = pack_frag(ap1, k0);
        v16bf b0 = pack_frag(bp0, k0);
        v16bf b1 = pack_frag(bp1, k0);
        v16bf b2 = pack_frag(bp2, k0);
        v16bf b3 = pack_frag(bp3, k0);
        acc[0][0] = __builtin_amdgcn_wmma_f32_16x16x32_bf16(false, a0, false, b0, (short)0, acc[0][0], false, false);
        acc[0][1] = __builtin_amdgcn_wmma_f32_16x16x32_bf16(false, a0, false, b1, (short)0, acc[0][1], false, false);
        acc[0][2] = __builtin_amdgcn_wmma_f32_16x16x32_bf16(false, a0, false, b2, (short)0, acc[0][2], false, false);
        acc[0][3] = __builtin_amdgcn_wmma_f32_16x16x32_bf16(false, a0, false, b3, (short)0, acc[0][3], false, false);
        acc[1][0] = __builtin_amdgcn_wmma_f32_16x16x32_bf16(false, a1, false, b0, (short)0, acc[1][0], false, false);
        acc[1][1] = __builtin_amdgcn_wmma_f32_16x16x32_bf16(false, a1, false, b1, (short)0, acc[1][1], false, false);
        acc[1][2] = __builtin_amdgcn_wmma_f32_16x16x32_bf16(false, a1, false, b2, (short)0, acc[1][2], false, false);
        acc[1][3] = __builtin_amdgcn_wmma_f32_16x16x32_bf16(false, a1, false, b3, (short)0, acc[1][3], false, false);
    }

    // C/D layout: VGPR v -> M = v (lanes 0-15) / v+8 (lanes 16-31); N = lane&15
    const int mbase = m0 + ((lane < 16) ? 0 : 8);
#pragma unroll
    for (int i = 0; i < 2; ++i) {
#pragma unroll
        for (int j = 0; j < 4; ++j) {
            const int n = n0 + 16 * j + r;
            const float bv = bias[n];
#pragma unroll
            for (int v = 0; v < 8; ++v)
                out[(size_t)(mbase + 16 * i + v) * ldo + n] = acc[i][j][v] + bv;
        }
    }
}

// ---------------------------------------------------------------------------
// Fused GRU gates: r,z,n + state update; writes both f32 and bf16 hidden.
// gi/gh already include b_ih/b_hh from the GEMM epilogues.
// ---------------------------------------------------------------------------
__global__ void gru_gates(const float* __restrict__ gi,
                          const float* __restrict__ gh,
                          float* __restrict__ h, __bf16* __restrict__ hb) {
    int i = blockIdx.x * blockDim.x + threadIdx.x;
    if (i >= BATCH * H) return;
    int b = i / H, j = i % H;
    const float* gib = gi + (size_t)b * G3;
    const float* ghb = gh + (size_t)b * G3;
    float rg = sigmoidf_(gib[j]         + ghb[j]);
    float zg = sigmoidf_(gib[H + j]     + ghb[H + j]);
    float ng = tanhf(gib[2 * H + j] + rg * ghb[2 * H + j]);
    float hn = (1.0f - zg) * ng + zg * h[i];
    h[i]  = hn;
    hb[i] = (__bf16)hn;
}

// ---------------------------------------------------------------------------
// Decoder projection: out[b,j] = inp[b,j] + h[b,:]·fc_w[j,:] + fc_b[j]
// Wave tile: 16Mx64N (full padded N); columns >= 55 masked on store.
// ---------------------------------------------------------------------------
__global__ void out_proj_wmma(const __bf16* __restrict__ hb,     // [B][H]
                              const __bf16* __restrict__ fcwb,   // [64][H]
                              const float* __restrict__ fcb,     // [55]
                              const float* __restrict__ inp, int inp_stride,
                              float* __restrict__ outp, int out_stride) {
    const int wave = threadIdx.x >> 5;
    const int lane = threadIdx.x & 31;
    const int tile = blockIdx.x * (blockDim.x >> 5) + wave;  // M tile index
    const int m0 = tile * 16;
    const int r  = lane & 15;
    const int kb = (lane < 16) ? 0 : 8;

    const __bf16* ap  = hb + (size_t)(m0 + r) * H + kb;
    const __bf16* bp0 = fcwb + (size_t)r * H + kb;
    const __bf16* bp1 = bp0 + (size_t)16 * H;
    const __bf16* bp2 = bp0 + (size_t)32 * H;
    const __bf16* bp3 = bp0 + (size_t)48 * H;

    v8f acc[4] = {};
#pragma unroll 1
    for (int k0 = 0; k0 < H; k0 += 32) {
        v16bf a  = pack_frag(ap, k0);
        v16bf b0 = pack_frag(bp0, k0);
        v16bf b1 = pack_frag(bp1, k0);
        v16bf b2 = pack_frag(bp2, k0);
        v16bf b3 = pack_frag(bp3, k0);
        acc[0] = __builtin_amdgcn_wmma_f32_16x16x32_bf16(false, a, false, b0, (short)0, acc[0], false, false);
        acc[1] = __builtin_amdgcn_wmma_f32_16x16x32_bf16(false, a, false, b1, (short)0, acc[1], false, false);
        acc[2] = __builtin_amdgcn_wmma_f32_16x16x32_bf16(false, a, false, b2, (short)0, acc[2], false, false);
        acc[3] = __builtin_amdgcn_wmma_f32_16x16x32_bf16(false, a, false, b3, (short)0, acc[3], false, false);
    }

    const int mbase = m0 + ((lane < 16) ? 0 : 8);
#pragma unroll
    for (int j = 0; j < 4; ++j) {
        const int n = 16 * j + r;
        if (n < IN) {
            const float bv = fcb[n];
#pragma unroll
            for (int v = 0; v < 8; ++v) {
                int row = mbase + v;
                outp[(size_t)row * out_stride + n] =
                    inp[(size_t)row * inp_stride + n] + acc[j][v] + bv;
            }
        }
    }
}

// ---------------------------------------------------------------------------
// Host orchestration (graph-capture safe: kernel launches only).
// ---------------------------------------------------------------------------
extern "C" void kernel_launch(void* const* d_in, const int* in_sizes, int n_in,
                              void* d_out, int out_size, void* d_ws, size_t ws_size,
                              hipStream_t stream) {
    const float* enc = (const float*)d_in[0];   // [B,50,55]
    const float* dec = (const float*)d_in[1];   // [B,25,55]
    const float* Wih = (const float*)d_in[2];   // [3072,55]
    const float* Whh = (const float*)d_in[3];   // [3072,1024]
    const float* bih = (const float*)d_in[4];   // [3072]
    const float* bhh = (const float*)d_in[5];   // [3072]
    const float* fcw = (const float*)d_in[6];   // [55,1024]
    const float* fcb = (const float*)d_in[7];   // [55]
    float* out = (float*)d_out;                 // [B,25,55]

    // Workspace carve-out (256B aligned)
    char* ws = (char*)d_ws;
    auto alloc = [&](size_t bytes) -> char* {
        char* p = ws;
        ws += (bytes + 255) & ~(size_t)255;
        return p;
    };
    __bf16* Whh_b = (__bf16*)alloc((size_t)G3 * H * 2);    //  6.0 MB
    __bf16* Wih_b = (__bf16*)alloc((size_t)G3 * KP * 2);   //  0.4 MB
    __bf16* fcw_b = (__bf16*)alloc((size_t)KP * H * 2);    //  0.13 MB
    float*  h_f32 = (float*) alloc((size_t)BATCH * H * 4); //  4.0 MB
    __bf16* h_b16 = (__bf16*)alloc((size_t)BATCH * H * 2); //  2.0 MB
    __bf16* x_b16 = (__bf16*)alloc((size_t)BATCH * KP * 2);// 0.13 MB
    float*  gi    = (float*) alloc((size_t)BATCH * G3 * 4);// 12.0 MB
    float*  gh    = (float*) alloc((size_t)BATCH * G3 * 4);// 12.0 MB

    prep_weights<<<512, 256, 0, stream>>>(Wih, Whh, fcw, Wih_b, Whh_b, fcw_b);
    init_h<<<(BATCH * H) / 256, 256, 0, stream>>>(h_f32, h_b16);

    const dim3 gblk(32 * WPB);                         // 4 waves / block
    const int tilesN4  = G3 / 64;                      // 48
    const int tilesM2  = BATCH / 32;                   // 32
    const int gemmGrid = tilesM2 * tilesN4 / WPB;      // 384 blocks
    const int projGrid = (BATCH / 16) / WPB;           // 16 blocks
    const int ewGrid   = (BATCH * H) / 256;
    const int cvGrid   = (BATCH * KP) / 256;

    // ----- Encoder: 49 GRU steps -----
    for (int t = 0; t < SRC_T - 1; ++t) {
        conv_x<<<cvGrid, 256, 0, stream>>>(enc + (size_t)t * IN, SRC_T * IN, x_b16);
        gemm_bf16_wmma<<<gemmGrid, gblk, 0, stream>>>(x_b16, KP, Wih_b, KP, bih,
                                                      gi, G3, tilesN4, KP);
        gemm_bf16_wmma<<<gemmGrid, gblk, 0, stream>>>(h_b16, H, Whh_b, H, bhh,
                                                      gh, G3, tilesN4, H);
        gru_gates<<<ewGrid, 256, 0, stream>>>(gi, gh, h_f32, h_b16);
    }

    // ----- Decoder: 25 autoregressive steps -----
    for (int t = 0; t < TGT_T; ++t) {
        const float* inp = (t == 0) ? dec : (out + (size_t)(t - 1) * IN);
        const int istride = TGT_T * IN;   // both dec and out are [B,25,55]
        conv_x<<<cvGrid, 256, 0, stream>>>(inp, istride, x_b16);
        gemm_bf16_wmma<<<gemmGrid, gblk, 0, stream>>>(x_b16, KP, Wih_b, KP, bih,
                                                      gi, G3, tilesN4, KP);
        gemm_bf16_wmma<<<gemmGrid, gblk, 0, stream>>>(h_b16, H, Whh_b, H, bhh,
                                                      gh, G3, tilesN4, H);
        gru_gates<<<ewGrid, 256, 0, stream>>>(gi, gh, h_f32, h_b16);
        out_proj_wmma<<<projGrid, gblk, 0, stream>>>(h_b16, fcw_b, fcb,
                                                     inp, istride,
                                                     out + (size_t)t * IN,
                                                     TGT_T * IN);
    }
}
